// MyRNN_9320079032909
// MI455X (gfx1250) — matrix operand order repaired and checked by reading, not compiled, not run on hardware
//
#include <hip/hip_runtime.h>
#include <stdint.h>

// Fully fused 3-layer GRU (B=4096, T=1024, I=1, H=16, G=48) + FC head.
//
// v2: the recurrence is latency-bound (256 waves, each on its own SIMD), so
// the LDS C->A transpose of v1 (3x ds round trips per step) is replaced by a
// register-only reformulation:
//   - Compute gates transposed: D[j,b] = W[j,:] . h[:,b]; A-operand = weights
//     (static, 16 frags resident in VGPRs), B-operand = hidden state
//     (columns = batch).
//   - WMMA C-layout output (lane=batch col, regs=j) -> next-step B-layout via
//     4x v_cvt_pk_f16_f32 + 4x v_permlanex16_b32 (half-wave swap). Zero LDS,
//     zero ds-waits in the whole T loop.
//   - B's spare K rows carry a 1.0 row (K16) and the x(t) row (K17): all
//     biases and the layer-0 input projection fold into the WMMAs for free.
// 16 v_wmma_f32_16x16x32_f16 per timestep, everything else pure VALU.

#define NW    4      // waves per block; each wave owns 16 batch rows
#define B_SZ  4096
#define T_SZ  1024

typedef __attribute__((ext_vector_type(16))) _Float16 v16h;
typedef __attribute__((ext_vector_type(8)))  float    v8f;

union VB  { v16h v; uint32_t u[8]; _Float16 h[16]; };
union F16P { _Float16 f[2]; uint32_t u; };
union FI  { float f; uint32_t u; };

__device__ __forceinline__ v8f wmma_f16(v16h a, v16h b, v8f c) {
  return __builtin_amdgcn_wmma_f32_16x16x32_f16(false, a, false, b,
                                                (short)0, c, false, false);
}

// swap the two 16-lane halves (identity within each half)
__device__ __forceinline__ uint32_t xswap16(uint32_t v) {
  return __builtin_amdgcn_permlanex16(v, v, 0x76543210u, 0xfedcba98u, false, false);
}

__device__ __forceinline__ float sigm(float x) {
  return __builtin_amdgcn_rcpf(1.0f + __builtin_amdgcn_exp2f(x * -1.44269504f));
}
__device__ __forceinline__ float tanh_fast(float x) {
  return 1.0f - 2.0f * __builtin_amdgcn_rcpf(1.0f + __builtin_amdgcn_exp2f(x * 2.88539008f));
}

// biases already inside the WMMA results (via the 1.0 row)
__device__ __forceinline__ v8f gru_gates(v8f ar, v8f az, v8f axn, v8f ahn, v8f h) {
  v8f o;
#pragma unroll
  for (int i = 0; i < 8; ++i) {
    float r = sigm(ar[i]);
    float z = sigm(az[i]);
    float n = tanh_fast(axn[i] + r * ahn[i]);
    o[i] = n + z * (h[i] - n);
  }
  return o;
}

// A fragment (16x32 f16), M = output index j = lane&15, contiguous K = 16*hi + half:
//   lanes 0..15: halves 0..15 = weight row [0 .. 15]
//   lanes 16..31: half0 = K16 (bias), half1 = K17 (layer-0 input weight)
//   lanes 16..31: halves 2..15 = K18..31 = 0
__device__ __forceinline__ v16h afrag(const float* wrow, float sLo, int hi,
                                      float k16, float k17) {
  const float* p = wrow;
  v16h a;
#pragma unroll
  for (int i = 0; i < 16; ++i) a[i] = hi ? (_Float16)0.f : (_Float16)(p[i] * sLo);
  if (hi) { a[0] = (_Float16)k16; a[1] = (_Float16)k17; }
  return a;
}

// Rebuild a B-state fragment from the new hidden state (WMMA C layout:
// lane holds column b = lane&15, regs hold j = r + 8*hi):
//   lanes 0..15 : halves 0..7 = own j (packed f16), halves 8..15 = swapped j
//   lanes 16..31: u[0] = static (K16=1.0, K17=x or 0), rest 0
__device__ __forceinline__ void rebuild_B(VB& b, v8f hnew, int hi, uint32_t st0) {
  uint32_t pk[4], sw[4];
#pragma unroll
  for (int p = 0; p < 4; ++p) {
    F16P q;
    q.f[0] = (_Float16)hnew[2 * p];
    q.f[1] = (_Float16)hnew[2 * p + 1];
    pk[p] = q.u;
  }
#pragma unroll
  for (int p = 0; p < 4; ++p) sw[p] = xswap16(pk[p]);
  b.u[0] = hi ? st0 : pk[0];
  b.u[1] = hi ? 0u : pk[1];
  b.u[2] = hi ? 0u : pk[2];
  b.u[3] = hi ? 0u : pk[3];
#pragma unroll
  for (int p = 0; p < 4; ++p) b.u[4 + p] = hi ? 0u : sw[p];
}

#define ONE_F16_LO 0x00003C00u   // half(1.0) in the K16 slot

__global__ __launch_bounds__(NW * 32, 1) void gru3_wmma(
    const float* __restrict__ x,      // [B, T]
    const float* __restrict__ w_ih0,  // [48]
    const float* __restrict__ w_ih12, // [2, 48, 16]
    const float* __restrict__ w_hh,   // [3, 48, 16]
    const float* __restrict__ b_ih,   // [3, 48]
    const float* __restrict__ b_hh,   // [3, 48]
    const float* __restrict__ fc_w,   // [5, 16]
    const float* __restrict__ fc_b,   // [5]
    float* __restrict__ out)          // [B, 5]
{
  const int lane = threadIdx.x & 31;
  const int hi   = lane >> 4;
  const int ln   = lane & 15;
  const int wave = threadIdx.x >> 5;
  const int base = (blockIdx.x * NW + wave) * 16;

  const float* whh0 = w_hh;
  const float* whh1 = w_hh + 48 * 16;
  const float* whh2 = w_hh + 2 * 48 * 16;
  const float* wih1 = w_ih12;
  const float* wih2 = w_ih12 + 48 * 16;

  // ---- static A fragments (weights + folded biases); 16 frags ----
  // layer 0 (input proj + bias fused via K16/K17 rows of B0)
  v16h A0r  = afrag(whh0 + (0  + ln) * 16, 1.f, hi, b_ih[ln]      + b_hh[ln],      w_ih0[ln]);
  v16h A0z  = afrag(whh0 + (16 + ln) * 16, 1.f, hi, b_ih[16 + ln] + b_hh[16 + ln], w_ih0[16 + ln]);
  v16h A0xn = afrag(whh0 + (32 + ln) * 16, 0.f, hi, b_ih[32 + ln],                 w_ih0[32 + ln]);
  v16h A0hn = afrag(whh0 + (32 + ln) * 16, 1.f, hi, b_hh[32 + ln],                 0.f);
  // layer 1
  v16h A1ir = afrag(wih1 + (0  + ln) * 16, 1.f, hi, b_ih[48 + ln] + b_hh[48 + ln], 0.f);
  v16h A1hr = afrag(whh1 + (0  + ln) * 16, 1.f, hi, 0.f, 0.f);
  v16h A1iz = afrag(wih1 + (16 + ln) * 16, 1.f, hi, b_ih[64 + ln] + b_hh[64 + ln], 0.f);
  v16h A1hz = afrag(whh1 + (16 + ln) * 16, 1.f, hi, 0.f, 0.f);
  v16h A1in = afrag(wih1 + (32 + ln) * 16, 1.f, hi, b_ih[80 + ln], 0.f);
  v16h A1hn = afrag(whh1 + (32 + ln) * 16, 1.f, hi, b_hh[80 + ln], 0.f);
  // layer 2
  v16h A2ir = afrag(wih2 + (0  + ln) * 16, 1.f, hi, b_ih[96 + ln]  + b_hh[96 + ln],  0.f);
  v16h A2hr = afrag(whh2 + (0  + ln) * 16, 1.f, hi, 0.f, 0.f);
  v16h A2iz = afrag(wih2 + (16 + ln) * 16, 1.f, hi, b_ih[112 + ln] + b_hh[112 + ln], 0.f);
  v16h A2hz = afrag(whh2 + (16 + ln) * 16, 1.f, hi, 0.f, 0.f);
  v16h A2in = afrag(wih2 + (32 + ln) * 16, 1.f, hi, b_ih[128 + ln], 0.f);
  v16h A2hn = afrag(whh2 + (32 + ln) * 16, 1.f, hi, b_hh[128 + ln], 0.f);

  // ---- recurrent state ----
  VB Bh0, Bh1, Bh2;      // [h ; 1.0 ; (x)] in B layout
#pragma unroll
  for (int p = 0; p < 8; ++p) {
    Bh0.u[p] = 0u; Bh1.u[p] = 0u; Bh2.u[p] = 0u;
  }
  Bh0.u[0] = hi ? ONE_F16_LO : 0u;
  Bh1.u[0] = hi ? ONE_F16_LO : 0u;
  Bh2.u[0] = hi ? ONE_F16_LO : 0u;

  v8f h0, h1, h2, zc;
#pragma unroll
  for (int i = 0; i < 8; ++i) { h0[i] = 0.f; h1[i] = 0.f; h2[i] = 0.f; zc[i] = 0.f; }

  const float4* xrow = reinterpret_cast<const float4*>(x + (size_t)(base + ln) * T_SZ);

#pragma unroll 1
  for (int tc = 0; tc < T_SZ / 4; ++tc) {
    float4 xv = xrow[tc];
#pragma unroll
    for (int s = 0; s < 4; ++s) {
      float xt = (s == 0) ? xv.x : (s == 1) ? xv.y : (s == 2) ? xv.z : xv.w;
      // x(t) into B0's K17 row (hi lanes, high half of dword 0)
      F16P xq; xq.f[0] = (_Float16)1.0f; xq.f[1] = (_Float16)xt;
      Bh0.u[0] = hi ? xq.u : Bh0.u[0];

      // ---- layer 0 (hidden + input proj + biases in one WMMA each) ----
      v8f ar  = wmma_f16(A0r,  Bh0.v, zc);
      v8f az  = wmma_f16(A0z,  Bh0.v, zc);
      v8f axn = wmma_f16(A0xn, Bh0.v, zc);   // b_ih_n + x*w_ih0_n
      v8f ahn = wmma_f16(A0hn, Bh0.v, zc);   // w_hh_n*h0 + b_hh_n
      h0 = gru_gates(ar, az, axn, ahn, h0);
      rebuild_B(Bh0, h0, hi, ONE_F16_LO);

      // ---- layer 1 ----
      ar  = wmma_f16(A1hr, Bh1.v, wmma_f16(A1ir, Bh0.v, zc));
      az  = wmma_f16(A1hz, Bh1.v, wmma_f16(A1iz, Bh0.v, zc));
      axn = wmma_f16(A1in, Bh0.v, zc);
      ahn = wmma_f16(A1hn, Bh1.v, zc);
      h1 = gru_gates(ar, az, axn, ahn, h1);
      rebuild_B(Bh1, h1, hi, ONE_F16_LO);

      // ---- layer 2 ----
      ar  = wmma_f16(A2hr, Bh2.v, wmma_f16(A2ir, Bh1.v, zc));
      az  = wmma_f16(A2hz, Bh2.v, wmma_f16(A2iz, Bh1.v, zc));
      axn = wmma_f16(A2in, Bh1.v, zc);
      ahn = wmma_f16(A2hn, Bh2.v, zc);
      h2 = gru_gates(ar, az, axn, ahn, h2);
      rebuild_B(Bh2, h2, hi, ONE_F16_LO);
    }
  }

  // ---- FC head on h2(T-1): lane holds j = 8*hi..8*hi+7 of column b = ln ----
#pragma unroll
  for (int o = 0; o < 5; ++o) {
    float part = 0.f;
#pragma unroll
    for (int r = 0; r < 8; ++r) part += fc_w[o * 16 + r + 8 * hi] * h2[r];
    FI pu, qu;
    pu.f = part;
    qu.u = xswap16(pu.u);
    float tot = part + qu.f + fc_b[o];
    if (hi == 0) out[(size_t)(base + ln) * 5 + o] = tot;
  }
}

extern "C" void kernel_launch(void* const* d_in, const int* in_sizes, int n_in,
                              void* d_out, int out_size, void* d_ws, size_t ws_size,
                              hipStream_t stream) {
  const float* x      = (const float*)d_in[0];
  const float* w_ih0  = (const float*)d_in[1];
  const float* w_ih12 = (const float*)d_in[2];
  const float* w_hh   = (const float*)d_in[3];
  const float* b_ih   = (const float*)d_in[4];
  const float* b_hh   = (const float*)d_in[5];
  const float* fc_w   = (const float*)d_in[6];
  const float* fc_b   = (const float*)d_in[7];
  float* out = (float*)d_out;

  dim3 grid(B_SZ / (NW * 16));   // 64 blocks
  dim3 block(NW * 32);           // 4 waves/block, 1 wave = 16 batch rows
  hipLaunchKernelGGL(gru3_wmma, grid, block, 0, stream,
                     x, w_ih0, w_ih12, w_hh, b_ih, b_hh, fc_w, fc_b, out);
}
